// TransformerLayer_1589137899794
// MI455X (gfx1250) — matrix operand order repaired and checked by reading, not compiled, not run on hardware
//
#include <hip/hip_runtime.h>
#include <hip/hip_bf16.h>
#include <math.h>

typedef __attribute__((ext_vector_type(16))) _Float16 v16h;
typedef __attribute__((ext_vector_type(8)))  _Float16 v8h;
typedef __attribute__((ext_vector_type(8)))  float    v8f;

#define DEV static __device__ __forceinline__

// ---- problem constants (B=2, H=128, W=256, D=128, NS=8) ----
constexpr int HGT  = 128;
constexpr int WID  = 256;
constexpr int DIM  = 128;
constexpr int LTOK = HGT * WID;          // 32768
constexpr int MROW = 2 * LTOK;           // 65536 total rows
constexpr int WINT = 512;                // tokens per window (16x32)
constexpr int FFH  = 1024;               // ffn hidden

// ---------- WMMA helpers ----------
DEV v8f wmma_f16(v16h a, v16h b, v8f c) {
  return __builtin_amdgcn_wmma_f32_16x16x32_f16(false, a, false, b, (short)0, c,
                                                false, false);
}

// A fragment: 16x32 f16, row-major source with leading dim ld.
// lane l: row = m0 + (l&15); elems 0..7 = K half*8..+8 ; elems 8..15 = K 16+half*8..+8
DEV v16h load_a(const _Float16* base, int ld, int m0, int k0) {
  int lane = threadIdx.x & 31;
  int m  = m0 + (lane & 15);
  int hf = (lane >> 4) << 3;                    // 0 or 8
  const _Float16* p = base + (long)m * ld + k0 + hf;
  v8h lo = *(const v8h*)p;
  v8h hi = *(const v8h*)(p + 16);
  v16h a;
#pragma unroll
  for (int i = 0; i < 8; i++) { a[i] = lo[i]; a[i + 8] = hi[i]; }
  return a;
}

// B fragment: 32x16 f16, from an N-major (transposed) matrix bt[N][ldk].
// lane l: col n = n0 + (l&15); elems 0..7 = K half*8..+8 ; elems 8..15 = K 16+half*8..+8
DEV v16h load_bt(const _Float16* bt, int ldk, int n0, int k0) {
  int lane = threadIdx.x & 31;
  int n  = n0 + (lane & 15);
  int kb = k0 + ((lane >> 4) << 3);
  const _Float16* p = bt + (long)n * ldk + kb;
  v8h lo = *(const v8h*)p;
  v8h hi = *(const v8h*)(p + 16);
  v16h b;
#pragma unroll
  for (int i = 0; i < 8; i++) { b[i] = lo[i]; b[i + 8] = hi[i]; }
  return b;
}

// async cache->LDS 16-byte copy (CDNA5 GLOBAL_LOAD_ASYNC_TO_LDS_B128, ASYNCcnt)
DEV void async_copy_b128(unsigned lds_byte_addr, const void* gaddr) {
  asm volatile("global_load_async_to_lds_b128 %0, %1, off"
               :: "v"(lds_byte_addr), "v"(gaddr) : "memory");
}
DEV void wait_asynccnt0() {
  asm volatile("s_wait_asynccnt 0x0" ::: "memory");
}
DEV unsigned lds_addr(const void* p) {        // generic LDS ptr -> wave LDS offset
  return (unsigned)(unsigned long long)p;     // flat LDS aperture: low 32 bits
}

// ---------- window <-> linear row mapping (shifted, Swin roll) ----------
// NOTE: for any 16-aligned row block these mappings are affine (base + idx):
// a 16-aligned run of 16 tokens shares h and never crosses a 32-wide window.
DEV int win_of_linear(int r) {              // linear token row -> windowed row
  int b = r >> 15;                          // / LTOK
  int rem = r & (LTOK - 1);
  int h = rem >> 8, w = rem & 255;
  int hh = (h + HGT - 8) & (HGT - 1);       // roll -sh
  int wr = (w + WID - 16) & (WID - 1);      // roll -sw
  int win = (((b << 3) + (hh >> 4)) << 3) + (wr >> 5);
  int t = ((hh & 15) << 5) + (wr & 31);
  return win * WINT + t;
}
DEV int linear_of_win(int rw) {             // windowed row -> linear token row
  int win = rw >> 9, t = rw & 511;
  int b = win >> 6, wb = win & 63;
  int hh = ((wb >> 3) << 4) + (t >> 5);
  int wr = ((wb & 7) << 5) + (t & 31);
  int h = (hh + 8) & (HGT - 1);
  int w = (wr + 16) & (WID - 1);
  return (b << 15) + (h << 8) + w;
}

// ---------- kernel 0: fp32 -> f16 activations ----------
__global__ void k_cast(const float* __restrict__ s, const float* __restrict__ t,
                       _Float16* __restrict__ s16, _Float16* __restrict__ t16) {
  long i = (long)blockIdx.x * blockDim.x + threadIdx.x;
  if (i < (long)MROW * DIM) {
    s16[i] = (_Float16)s[i];
    t16[i] = (_Float16)t[i];
  }
}

// ---------- kernel 0b: fp32 weight [K][N] -> f16 transposed [N][K] ----------
__global__ void k_transpose(const float* __restrict__ w, _Float16* __restrict__ wt,
                            int K, int N) {
  int i = blockIdx.x * blockDim.x + threadIdx.x;
  if (i < K * N) {
    int k = i / N, n = i % N;
    wt[(long)n * K + k] = (_Float16)w[i];
  }
}

// ---------- kernel 1: fused Q/K/V projection, scatter to windowed layout ----------
__global__ void k_qkv(const _Float16* __restrict__ s16, const _Float16* __restrict__ t16,
                      const _Float16* __restrict__ wqt, const _Float16* __restrict__ wkt,
                      const _Float16* __restrict__ wvt,
                      _Float16* __restrict__ qw, _Float16* __restrict__ kw,
                      _Float16* __restrict__ vw) {
  int which = blockIdx.z;
  const _Float16* x  = (which == 0) ? s16 : t16;
  const _Float16* wt = (which == 0) ? wqt : (which == 1 ? wkt : wvt);
  _Float16* o        = (which == 0) ? qw  : (which == 1 ? kw  : vw);
  int m0 = blockIdx.x * 16, n0 = blockIdx.y * 16;
  v8f c = {0, 0, 0, 0, 0, 0, 0, 0};
#pragma unroll
  for (int kc = 0; kc < 4; kc++) {
    v16h a = load_a(x, DIM, m0, kc * 32);
    v16h b = load_bt(wt, DIM, n0, kc * 32);
    c = wmma_f16(a, b, c);
  }
  int lane = threadIdx.x & 31, n = n0 + (lane & 15), hf = lane >> 4;
  int rw0 = win_of_linear(m0);              // uniform affine base (SALU)
#pragma unroll
  for (int j = 0; j < 8; j++) {
    int rw = rw0 + j + (hf << 3);
    o[(long)rw * DIM + n] = (_Float16)c[j];
  }
}

// ---------- kernel 2: windowed masked attention (flash-style, 8 waves/WG) ----------
__global__ void __launch_bounds__(256)
k_attn(const _Float16* __restrict__ qw, const _Float16* __restrict__ kw,
       const _Float16* __restrict__ vw, const float* __restrict__ mask,
       _Float16* __restrict__ aw) {
  __shared__ __align__(32) _Float16 Kt[32 * 128];   // K tile, row-major [key][d]
  __shared__ __align__(32) _Float16 Vr[32 * 128];   // V tile, row-major (async landing)
  __shared__ __align__(32) _Float16 Vt[128 * 32];   // V tile transposed [d][key]
  __shared__ __align__(32) _Float16 Pb[8][16 * 32]; // per-wave P scratch

  const int wave = threadIdx.x >> 5, lane = threadIdx.x & 31;
  const int n = lane & 15, hf = lane >> 4;
  const int win = blockIdx.y;
  const int qtile = blockIdx.x * 8 + wave;
  const int m0 = qtile * 16;                        // query rows within window
  const _Float16* qbase = qw + (long)win * WINT * DIM;
  const float* mbase = mask + (long)(win & 63) * WINT * WINT;

  v16h aq[4];
#pragma unroll
  for (int c = 0; c < 4; c++) aq[c] = load_a(qbase, DIM, m0, c * 32);

  const v8f vz = {0, 0, 0, 0, 0, 0, 0, 0};
  v8f o[8];
  float mrow[8], lrow[8];
#pragma unroll
  for (int j = 0; j < 8; j++) { o[j] = vz; mrow[j] = -1e30f; lrow[j] = 0.f; }

  const float sc = 0.08838834764831845f;            // 1/sqrt(128)
  const int t = threadIdx.x;
  const int row = t >> 3, seg = t & 7;              // 32B per thread per tile

  for (int kt = 0; kt < 16; kt++) {
    __syncthreads();                                // prior LDS consumers done
    {   // async cache->LDS staging of K and V tiles (ASYNCcnt path)
      long gr = (long)(win * WINT + kt * 32 + row) * DIM + seg * 16;
      unsigned lk = lds_addr(&Kt[row * 128 + seg * 16]);
      unsigned lv = lds_addr(&Vr[row * 128 + seg * 16]);
      async_copy_b128(lk,      kw + gr);
      async_copy_b128(lk + 16, kw + gr + 8);
      async_copy_b128(lv,      vw + gr);
      async_copy_b128(lv + 16, vw + gr + 8);
      if (kt + 1 < 16) {   // warm L2 for next tile (global_prefetch_b8)
        __builtin_prefetch(kw + gr + 32 * DIM, 0, 1);
        __builtin_prefetch(vw + gr + 32 * DIM, 0, 1);
      }
      wait_asynccnt0();
    }
    __syncthreads();
    {   // LDS->LDS transpose of V tile: Vt[d][kk] = Vr[kk][d]
#pragma unroll
      for (int i = 0; i < 16; i++) {
        int idx = t * 16 + i;
        int d = idx >> 5, kk = idx & 31;
        Vt[d * 32 + kk] = Vr[kk * 128 + d];
      }
    }
    __syncthreads();

    v8f s0 = vz, s1 = vz;
#pragma unroll
    for (int c = 0; c < 4; c++) {
      s0 = wmma_f16(aq[c], load_bt(Kt, 128, 0, c * 32), s0);
      s1 = wmma_f16(aq[c], load_bt(Kt, 128, 16, c * 32), s1);
    }

#pragma unroll
    for (int j = 0; j < 8; j++) {
      int qr = m0 + j + (hf << 3);
      int key = kt * 32;
      float v0 = s0[j] * sc + mbase[(long)qr * WINT + key + n];
      float v1 = s1[j] * sc + mbase[(long)qr * WINT + key + 16 + n];
      float mx = fmaxf(v0, v1);
#pragma unroll
      for (int d = 1; d < 16; d <<= 1) mx = fmaxf(mx, __shfl_xor(mx, d, 32));
      float mnew = fmaxf(mrow[j], mx);
      float fac = __expf(mrow[j] - mnew);
      float p0 = __expf(v0 - mnew), p1 = __expf(v1 - mnew);
      float rs = p0 + p1;
#pragma unroll
      for (int d = 1; d < 16; d <<= 1) rs += __shfl_xor(rs, d, 32);
      lrow[j] = lrow[j] * fac + rs;
      mrow[j] = mnew;
#pragma unroll
      for (int nt = 0; nt < 8; nt++) o[nt][j] *= fac;
      Pb[wave][(j + (hf << 3)) * 32 + n]      = (_Float16)p0;
      Pb[wave][(j + (hf << 3)) * 32 + 16 + n] = (_Float16)p1;
    }
    asm volatile("s_wait_dscnt 0" ::: "memory");    // P stores visible to own wave
    v16h ap = load_a(&Pb[wave][0], 32, 0, 0);
#pragma unroll
    for (int nt = 0; nt < 8; nt++)
      o[nt] = wmma_f16(ap, load_bt(Vt, 32, nt * 16, 0), o[nt]);
  }

#pragma unroll
  for (int j = 0; j < 8; j++) {
    float inv = 1.0f / lrow[j];
    long row8 = (long)win * WINT + m0 + j + (hf << 3);
#pragma unroll
    for (int nt = 0; nt < 8; nt++)
      aw[row8 * DIM + nt * 16 + n] = (_Float16)(o[nt][j] * inv);
  }
}

// ---------- kernel 3: message @ wm, LayerNorm, scatter back to linear order ----------
__global__ void k_msgproj(const _Float16* __restrict__ aw, const _Float16* __restrict__ wmt,
                          const float* __restrict__ g1, const float* __restrict__ b1,
                          _Float16* __restrict__ msg) {
  int m0 = blockIdx.x * 16;                         // windowed row block
  const v8f vz = {0, 0, 0, 0, 0, 0, 0, 0};
  v8f c[8];
#pragma unroll
  for (int i = 0; i < 8; i++) c[i] = vz;
#pragma unroll
  for (int kc = 0; kc < 4; kc++) {
    v16h a = load_a(aw, DIM, m0, kc * 32);
#pragma unroll
    for (int nt = 0; nt < 8; nt++)
      c[nt] = wmma_f16(a, load_bt(wmt, DIM, nt * 16, kc * 32), c[nt]);
  }
  int lane = threadIdx.x & 31, n = lane & 15, hf = lane >> 4;
  int lin0 = linear_of_win(m0);                     // uniform affine base (SALU)
#pragma unroll
  for (int j = 0; j < 8; j++) {
    float sum = 0.f;
#pragma unroll
    for (int nt = 0; nt < 8; nt++) sum += c[nt][j];
#pragma unroll
    for (int d = 1; d < 16; d <<= 1) sum += __shfl_xor(sum, d, 32);
    float mean = sum * (1.0f / DIM);
    float var = 0.f;
#pragma unroll
    for (int nt = 0; nt < 8; nt++) { float dv = c[nt][j] - mean; var += dv * dv; }
#pragma unroll
    for (int d = 1; d < 16; d <<= 1) var += __shfl_xor(var, d, 32);
    float rstd = rsqrtf(var * (1.0f / DIM) + 1e-5f);
    int lin = lin0 + j + (hf << 3);
#pragma unroll
    for (int nt = 0; nt < 8; nt++) {
      int col = nt * 16 + n;
      float y = (c[nt][j] - mean) * rstd * g1[col] + b1[col];
      msg[(long)lin * DIM + col] = (_Float16)y;
    }
  }
}

// ---------- kernel 4: FFN1 = gelu(concat(src,msg) @ W1) ----------
__global__ void k_ffn1(const _Float16* __restrict__ s16, const _Float16* __restrict__ msg,
                       const _Float16* __restrict__ w1t, _Float16* __restrict__ hdn) {
  int m0 = blockIdx.x * 16, n0 = blockIdx.y * 16;
  v8f c = {0, 0, 0, 0, 0, 0, 0, 0};
#pragma unroll
  for (int kc = 0; kc < 8; kc++) {
    const _Float16* x = (kc < 4) ? s16 : msg;
    int k0 = (kc < 4) ? kc * 32 : (kc - 4) * 32;
    v16h a = load_a(x, DIM, m0, k0);
    v16h b = load_bt(w1t, 256, n0, kc * 32);        // w1t: [1024][256]
    c = wmma_f16(a, b, c);
  }
  int lane = threadIdx.x & 31, n = n0 + (lane & 15), hf = lane >> 4;
#pragma unroll
  for (int j = 0; j < 8; j++) {
    long row = m0 + j + (hf << 3);
    float x = c[j];
    float g = 0.5f * x * (1.0f + erff(x * 0.70710678118654752f));
    hdn[row * FFH + n] = (_Float16)g;
  }
}

// ---------- kernel 5: FFN2 + LayerNorm + residual ----------
__global__ void k_ffn2(const _Float16* __restrict__ hdn, const _Float16* __restrict__ w2t,
                       const float* __restrict__ g2, const float* __restrict__ b2,
                       const float* __restrict__ src, float* __restrict__ out) {
  int m0 = blockIdx.x * 16;
  const v8f vz = {0, 0, 0, 0, 0, 0, 0, 0};
  v8f c[8];
#pragma unroll
  for (int i = 0; i < 8; i++) c[i] = vz;
  for (int kc = 0; kc < 32; kc++) {
    v16h a = load_a(hdn, FFH, m0, kc * 32);
#pragma unroll
    for (int nt = 0; nt < 8; nt++)
      c[nt] = wmma_f16(a, load_bt(w2t, FFH, nt * 16, kc * 32), c[nt]);
  }
  int lane = threadIdx.x & 31, n = lane & 15, hf = lane >> 4;
#pragma unroll
  for (int j = 0; j < 8; j++) {
    float sum = 0.f;
#pragma unroll
    for (int nt = 0; nt < 8; nt++) sum += c[nt][j];
#pragma unroll
    for (int d = 1; d < 16; d <<= 1) sum += __shfl_xor(sum, d, 32);
    float mean = sum * (1.0f / DIM);
    float var = 0.f;
#pragma unroll
    for (int nt = 0; nt < 8; nt++) { float dv = c[nt][j] - mean; var += dv * dv; }
#pragma unroll
    for (int d = 1; d < 16; d <<= 1) var += __shfl_xor(var, d, 32);
    float rstd = rsqrtf(var * (1.0f / DIM) + 1e-5f);
    long row = m0 + j + (hf << 3);
#pragma unroll
    for (int nt = 0; nt < 8; nt++) {
      int col = nt * 16 + n;
      float y = (c[nt][j] - mean) * rstd * g2[col] + b2[col];
      out[row * DIM + col] = src[row * DIM + col] + y;
    }
  }
}

// ---------- host-side launch ----------
extern "C" void kernel_launch(void* const* d_in, const int* in_sizes, int n_in,
                              void* d_out, int out_size, void* d_ws, size_t ws_size,
                              hipStream_t stream) {
  const float* source = (const float*)d_in[0];
  const float* target = (const float*)d_in[1];
  const float* wq = (const float*)d_in[2];
  const float* wk = (const float*)d_in[3];
  const float* wv = (const float*)d_in[4];
  const float* wm = (const float*)d_in[5];
  const float* g1 = (const float*)d_in[6];
  const float* b1 = (const float*)d_in[7];
  const float* w1 = (const float*)d_in[8];
  const float* w2 = (const float*)d_in[9];
  const float* g2 = (const float*)d_in[10];
  const float* b2 = (const float*)d_in[11];
  const float* mask = (const float*)d_in[12];
  float* out = (float*)d_out;

  char* ws = (char*)d_ws;
  size_t off = 0;
  auto take = [&](size_t bytes) -> char* {
    char* p = ws + off;
    off += (bytes + 255) & ~(size_t)255;
    return p;
  };
  const size_t actBytes = (size_t)MROW * DIM * sizeof(_Float16);
  _Float16* s16   = (_Float16*)take(actBytes);
  _Float16* t16   = (_Float16*)take(actBytes);
  _Float16* wqt   = (_Float16*)take((size_t)DIM * DIM * 2);
  _Float16* wkt   = (_Float16*)take((size_t)DIM * DIM * 2);
  _Float16* wvt   = (_Float16*)take((size_t)DIM * DIM * 2);
  _Float16* wmt   = (_Float16*)take((size_t)DIM * DIM * 2);
  _Float16* w1t   = (_Float16*)take((size_t)256 * FFH * 2);
  _Float16* w2t   = (_Float16*)take((size_t)FFH * DIM * 2);
  _Float16* qwin  = (_Float16*)take(actBytes);
  _Float16* kwin  = (_Float16*)take(actBytes);
  _Float16* vwin  = (_Float16*)take(actBytes);
  _Float16* awin  = (_Float16*)take(actBytes);
  _Float16* msg16 = (_Float16*)take(actBytes);
  _Float16* hdn16 = (_Float16*)take((size_t)MROW * FFH * sizeof(_Float16));
  (void)ws_size; (void)in_sizes; (void)n_in; (void)out_size;

  // 0) precision conversion / weight transposes
  k_cast<<<(MROW * DIM + 255) / 256, 256, 0, stream>>>(source, target, s16, t16);
  k_transpose<<<(DIM * DIM + 255) / 256, 256, 0, stream>>>(wq, wqt, DIM, DIM);
  k_transpose<<<(DIM * DIM + 255) / 256, 256, 0, stream>>>(wk, wkt, DIM, DIM);
  k_transpose<<<(DIM * DIM + 255) / 256, 256, 0, stream>>>(wv, wvt, DIM, DIM);
  k_transpose<<<(DIM * DIM + 255) / 256, 256, 0, stream>>>(wm, wmt, DIM, DIM);
  k_transpose<<<(256 * FFH + 255) / 256, 256, 0, stream>>>(w1, w1t, 256, FFH);
  k_transpose<<<(FFH * DIM + 255) / 256, 256, 0, stream>>>(w2, w2t, FFH, DIM);

  // 1) Q/K/V projections into windowed token order
  k_qkv<<<dim3(MROW / 16, DIM / 16, 3), 32, 0, stream>>>(s16, t16, wqt, wkt, wvt,
                                                         qwin, kwin, vwin);
  // 2) shifted-window masked attention (async cache->LDS staging)
  k_attn<<<dim3(4, 128), 256, 0, stream>>>(qwin, kwin, vwin, mask, awin);
  // 3) output projection + LN, back to linear order
  k_msgproj<<<MROW / 16, 32, 0, stream>>>(awin, wmt, g1, b1, msg16);
  // 4) FFN up-projection with concat + GELU
  k_ffn1<<<dim3(MROW / 16, FFH / 16), 32, 0, stream>>>(s16, msg16, w1t, hdn16);
  // 5) FFN down-projection + LN + residual
  k_ffn2<<<MROW / 16, 32, 0, stream>>>(hdn16, w2t, g2, b2, source, out);
}
